// ShiftedWindowAttention_13649406067198
// MI455X (gfx1250) — matrix operand order repaired and checked by reading, not compiled, not run on hardware
//
#include <hip/hip_runtime.h>
#include <hip/hip_bf16.h>

// ---------------------------------------------------------------------------
// Fused Swin shifted-window attention for MI455X (gfx1250).
//  * Big GEMMs (QKV, proj): fp32-accurate split-precision f16x3 emulation on
//    v_wmma_f32_16x16x32_f16 (hi/lo f16 decomposition, 3 WMMAs per K=32).
//  * Attention GEMMs (QK^T, P@V): native v_wmma_f32_16x16x4_f32.
// One 256-thread workgroup (8 wave32) = one 7x7 window, all staged in LDS
// (~286 KB of the 320 KB/WGP). Weights pre-split to f16 hi/lo in d_ws (1 MB,
// L2-resident across all 2048 blocks).
// ---------------------------------------------------------------------------

#define BATCH   32
#define IMG     56
#define CC      256
#define NH      8
#define HD      32
#define NT      49        // tokens per window
#define NWIN    8
#define SHIFTV  3

// f32 LDS strides / offsets (in floats)
#define QSTR    772       // qkv row stride (pad vs 64-bank)
#define SSTR    66        // score row stride
#define OFF_QKV 0
#define OFF_S   (NT * QSTR)              // 37828, overlaps the f16 x region
#define S_HEAD  (64 * SSTR)
#define SMEM_FLOATS (NT * QSTR + NH * S_HEAD)   // 71620 floats = 286480 B

// f16 LDS regions (offsets in halfs within the same smem block)
#define X16STR  264                       // half row stride (16B aligned)
#define XH_OFF  (2 * OFF_S)               // 75656: x hi, overlaps S region
#define XL_OFF  (XH_OFF + 64 * X16STR)    // x lo
#define OH_OFF  0                         // attn-out hi, reuses QKV region
#define OL_OFF  (64 * X16STR)             // attn-out lo

// d_ws half-array offsets
#define WQ_N    (768 * 256)
#define WP_N    (256 * 256)
#define WS_QH   0
#define WS_QL   (WQ_N)
#define WS_PH   (2 * WQ_N)
#define WS_PL   (2 * WQ_N + WP_N)
#define WS_TOT  (2 * WQ_N + 2 * WP_N)     // 524288 halfs = 1 MB

typedef __attribute__((ext_vector_type(2)))  float    v2f;
typedef __attribute__((ext_vector_type(8)))  float    v8f;
typedef __attribute__((ext_vector_type(8)))  _Float16 v8h;
typedef __attribute__((ext_vector_type(16))) _Float16 v16h;

__device__ __forceinline__ v8f wmma_f32x4(v2f a, v2f b, v8f c) {
    return __builtin_amdgcn_wmma_f32_16x16x4_f32(
        false, a, false, b, (short)0, c, false, false);
}
__device__ __forceinline__ v8f wmma_f16(v16h a, v16h b, v8f c) {
    return __builtin_amdgcn_wmma_f32_16x16x32_f16(
        false, a, false, b, (short)0, c, false, false);
}
__device__ __forceinline__ v16h cat16(v8h a, v8h b) {
    return __builtin_shufflevector(a, b, 0, 1, 2, 3, 4, 5, 6, 7,
                                         8, 9, 10, 11, 12, 13, 14, 15);
}
__device__ __forceinline__ int region3(int p) {
    return (p < 49) ? 0 : ((p < 53) ? 1 : 2);
}

// ---------------- pre-pass: split fp32 weights into f16 hi/lo ---------------
__global__ __launch_bounds__(256)
void split_weights_kernel(const float* __restrict__ qkvW,
                          const float* __restrict__ projW,
                          _Float16* __restrict__ ws) {
    int i = blockIdx.x * 256 + threadIdx.x;
    if (i < WQ_N) {
        float v = qkvW[i];
        _Float16 hv = (_Float16)v;
        ws[WS_QH + i] = hv;
        ws[WS_QL + i] = (_Float16)(v - (float)hv);
    }
    if (i < WP_N) {
        float v = projW[i];
        _Float16 hv = (_Float16)v;
        ws[WS_PH + i] = hv;
        ws[WS_PL + i] = (_Float16)(v - (float)hv);
    }
}

// ---------------------------- fused main kernel -----------------------------
__global__ __launch_bounds__(256)
void swin_window_attn_kernel(const float* __restrict__ x,
                             const float* __restrict__ qkvB,
                             const float* __restrict__ projB,
                             const float* __restrict__ relT,
                             const _Float16* __restrict__ w16,
                             float* __restrict__ out) {
    __shared__ __align__(16) float smem[SMEM_FLOATS];
    _Float16* smemh = (_Float16*)smem;

    const int tid   = threadIdx.x;
    const int lane  = tid & 31;
    const int wave  = tid >> 5;
    const int lmod  = lane & 15;
    const int lhalf = lane >> 4;

    const int widx = blockIdx.x;
    const int b    = widx >> 6;
    const int w64  = widx & 63;
    const int wi   = w64 >> 3;
    const int wj   = w64 & 7;

    // ---- phase 1: gather rolled window, split into f16 hi/lo in LDS -------
    for (int i = tid; i < NT * CC; i += 256) {
        int tok = i >> 8;
        int ch  = i & 255;
        int mr  = tok / 7, mc = tok - mr * 7;
        int sh  = (wi * 7 + mr + SHIFTV) % IMG;
        int sw  = (wj * 7 + mc + SHIFTV) % IMG;
        float v = x[(((size_t)b * IMG + sh) * IMG + sw) * CC + ch];
        _Float16 hv = (_Float16)v;
        smemh[XH_OFF + tok * X16STR + ch] = hv;
        smemh[XL_OFF + tok * X16STR + ch] = (_Float16)(v - (float)hv);
    }
    __syncthreads();

    // ---- phase 2: QKV gemm [49x256]x[256x768], f16x3 split precision ------
    const _Float16* qWh = w16 + WS_QH;
    const _Float16* qWl = w16 + WS_QL;
    for (int tn = 0; tn < 6; ++tn) {
        const int ncol = wave * 96 + tn * 16 + lmod;
        v8f acc[4];
#pragma unroll
        for (int t = 0; t < 4; ++t) acc[t] = (v8f){0.f,0.f,0.f,0.f,0.f,0.f,0.f,0.f};
        for (int k0 = 0; k0 < CC; k0 += 32) {
            const int bofs = ncol * CC + k0 + 16 * lhalf;
            v16h bh = *(const v16h*)&qWh[bofs];
            v16h bl = *(const v16h*)&qWl[bofs];
#pragma unroll
            for (int tm = 0; tm < 4; ++tm) {
                const int abase = (tm * 16 + lmod) * X16STR + k0 + 16 * lhalf;
                v16h ah = cat16(*(const v8h*)&smemh[XH_OFF + abase],
                                *(const v8h*)&smemh[XH_OFF + abase + 8]);
                v16h al = cat16(*(const v8h*)&smemh[XL_OFF + abase],
                                *(const v8h*)&smemh[XL_OFF + abase + 8]);
                acc[tm] = wmma_f16(ah, bh, acc[tm]);
                acc[tm] = wmma_f16(ah, bl, acc[tm]);
                acc[tm] = wmma_f16(al, bh, acc[tm]);
            }
        }
        const float bias = qkvB[ncol];
#pragma unroll
        for (int tm = 0; tm < 4; ++tm)
#pragma unroll
            for (int r = 0; r < 8; ++r) {
                int row = tm * 16 + r + lhalf * 8;
                if (row < NT)
                    smem[OFF_QKV + row * QSTR + ncol] = acc[tm][r] + bias;
            }
    }
    __syncthreads();

    // ---- phase 3: per-head attention (wave == head), native f32 WMMA ------
    const int h = wave;
    const int qoff = h * HD;
    const int koff = CC + h * HD;
    const int voff = 2 * CC + h * HD;
    float* Sh = &smem[OFF_S + h * S_HEAD];
    const float scale = 0.17677669529663687f;   // 1/sqrt(32)

    for (int tm = 0; tm < 4; ++tm) {
        const int arow = tm * 16 + lmod;
        for (int tn = 0; tn < 4; ++tn) {
            const int bcol = tn * 16 + lmod;
            v8f acc = {0.f, 0.f, 0.f, 0.f, 0.f, 0.f, 0.f, 0.f};
            for (int k0 = 0; k0 < HD; k0 += 4) {
                v2f a = *(const v2f*)&smem[OFF_QKV + arow * QSTR + qoff + k0 + 2 * lhalf];
                a = a * scale;
                v2f bb = *(const v2f*)&smem[OFF_QKV + bcol * QSTR + koff + k0 + 2 * lhalf];
                acc = wmma_f32x4(a, bb, acc);
            }
#pragma unroll
            for (int r = 0; r < 8; ++r) {
                int row = tm * 16 + r + lhalf * 8;
                int col = tn * 16 + lmod;
                float sv;
                if (col < NT) {
                    int rowc = row < NT ? row : NT - 1;
                    int mr = rowc / 7, mc = rowc - mr * 7;
                    int nr = col / 7,  nc = col - nr * 7;
                    int idx = (mr - nr + 6) * 13 + (mc - nc + 6);
                    float rb = relT[idx * NH + h];
                    int cm = region3(wi * 7 + mr) * 3 + region3(wj * 7 + mc);
                    int cn = region3(wi * 7 + nr) * 3 + region3(wj * 7 + nc);
                    float mv = (cm != cn) ? -100.0f : 0.0f;
                    sv = acc[r] + rb + mv;
                } else {
                    sv = -1e30f;
                }
                Sh[row * SSTR + col] = sv;
            }
        }
    }
    __syncthreads();

    // softmax (valid rows); zero pad cols so they vanish in P@V
    for (int m = lane; m < NT; m += 32) {
        float* rowp = Sh + m * SSTR;
        float mx = -3.0e38f;
        for (int c = 0; c < NT; ++c) mx = fmaxf(mx, rowp[c]);
        float sum = 0.f;
        for (int c = 0; c < NT; ++c) {
            float e = __expf(rowp[c] - mx);
            sum += e;
            rowp[c] = e;
        }
        float inv = 1.0f / sum;
        for (int c = 0; c < NT; ++c) rowp[c] *= inv;
        for (int c = NT; c < 64; ++c) rowp[c] = 0.f;
    }
    __syncthreads();

    // O = P @ V   (registers until all waves stop reading V)
    v8f Oacc[4][2];
#pragma unroll
    for (int tm = 0; tm < 4; ++tm) {
#pragma unroll
        for (int tnn = 0; tnn < 2; ++tnn) {
            v8f acc = {0.f, 0.f, 0.f, 0.f, 0.f, 0.f, 0.f, 0.f};
            const int arow = tm * 16 + lmod;
            const int feat = voff + tnn * 16 + lmod;
            for (int k0 = 0; k0 < 64; k0 += 4) {
                v2f a = *(const v2f*)&Sh[arow * SSTR + k0 + 2 * lhalf];
                int t0 = k0 + 2 * lhalf;
                int t1 = t0 + 1;
                t0 = t0 < NT ? t0 : NT - 1;
                t1 = t1 < NT ? t1 : NT - 1;
                v2f bb;
                bb.x = smem[OFF_QKV + t0 * QSTR + feat];
                bb.y = smem[OFF_QKV + t1 * QSTR + feat];
                acc = wmma_f32x4(a, bb, acc);
            }
            Oacc[tm][tnn] = acc;
        }
    }
    __syncthreads();   // everyone done with V before QKV region is reused

    // store O as f16 hi/lo into the (dead) QKV region
#pragma unroll
    for (int tm = 0; tm < 4; ++tm)
#pragma unroll
        for (int tnn = 0; tnn < 2; ++tnn)
#pragma unroll
            for (int r = 0; r < 8; ++r) {
                int row = tm * 16 + r + lhalf * 8;
                if (row < NT) {
                    float v = Oacc[tm][tnn][r];
                    _Float16 hv = (_Float16)v;
                    int o = row * X16STR + h * HD + tnn * 16 + lmod;
                    smemh[OH_OFF + o] = hv;
                    smemh[OL_OFF + o] = (_Float16)(v - (float)hv);
                }
            }
    __syncthreads();

    // ---- phase 4: proj gemm (f16x3) + reverse-roll scatter ----------------
    const _Float16* pWh = w16 + WS_PH;
    const _Float16* pWl = w16 + WS_PL;
    for (int tnn = 0; tnn < 2; ++tnn) {
        const int ncol = wave * 32 + tnn * 16 + lmod;
        v8f acc[4];
#pragma unroll
        for (int t = 0; t < 4; ++t) acc[t] = (v8f){0.f,0.f,0.f,0.f,0.f,0.f,0.f,0.f};
        for (int k0 = 0; k0 < CC; k0 += 32) {
            const int bofs = ncol * CC + k0 + 16 * lhalf;
            v16h bh = *(const v16h*)&pWh[bofs];
            v16h bl = *(const v16h*)&pWl[bofs];
#pragma unroll
            for (int tm = 0; tm < 4; ++tm) {
                const int abase = (tm * 16 + lmod) * X16STR + k0 + 16 * lhalf;
                v16h ah = cat16(*(const v8h*)&smemh[OH_OFF + abase],
                                *(const v8h*)&smemh[OH_OFF + abase + 8]);
                v16h al = cat16(*(const v8h*)&smemh[OL_OFF + abase],
                                *(const v8h*)&smemh[OL_OFF + abase + 8]);
                acc[tm] = wmma_f16(ah, bh, acc[tm]);
                acc[tm] = wmma_f16(ah, bl, acc[tm]);
                acc[tm] = wmma_f16(al, bh, acc[tm]);
            }
        }
        const float pb = projB[ncol];
#pragma unroll
        for (int tm = 0; tm < 4; ++tm)
#pragma unroll
            for (int r = 0; r < 8; ++r) {
                int row = tm * 16 + r + lhalf * 8;
                if (row < NT) {
                    int mr = row / 7, mc = row - mr * 7;
                    int oh = (wi * 7 + mr + SHIFTV) % IMG;
                    int ow = (wj * 7 + mc + SHIFTV) % IMG;
                    out[(((size_t)b * IMG + oh) * IMG + ow) * CC + ncol] =
                        acc[tm][r] + pb;
                }
            }
    }
}

extern "C" void kernel_launch(void* const* d_in, const int* in_sizes, int n_in,
                              void* d_out, int out_size, void* d_ws, size_t ws_size,
                              hipStream_t stream) {
    (void)in_sizes; (void)n_in; (void)ws_size; (void)out_size;
    const float* x      = (const float*)d_in[0];
    const float* qkvW   = (const float*)d_in[1];
    const float* qkvB   = (const float*)d_in[2];
    const float* projW  = (const float*)d_in[3];
    const float* projB  = (const float*)d_in[4];
    const float* relT   = (const float*)d_in[5];
    float* out = (float*)d_out;
    _Float16* w16 = (_Float16*)d_ws;

    // pre-pass: split weights into f16 hi/lo (1 MB in d_ws, L2-resident)
    split_weights_kernel<<<dim3((WQ_N + 255) / 256), dim3(256), 0, stream>>>(
        qkvW, projW, w16);

    swin_window_attn_kernel<<<dim3(BATCH * NWIN * NWIN), dim3(256), 0, stream>>>(
        x, qkvB, projB, relT, w16, out);
}
